// ASSAAttention_4148938408570
// MI455X (gfx1250) — matrix operand order repaired and checked by reading, not compiled, not run on hardware
//
#include <hip/hip_runtime.h>
#include <math.h>

// ---- CDNA5 WMMA types ----
typedef _Float16 v16h __attribute__((ext_vector_type(16)));
typedef _Float16 v8h  __attribute__((ext_vector_type(8)));
typedef float    v8f  __attribute__((ext_vector_type(8)));
typedef _Float16 h2   __attribute__((ext_vector_type(2)));

// Problem constants (reference: B=2, L=2048, H=16, E=64)
constexpr int BATCH = 2;
constexpr int LQ    = 2048;
constexpr int NH    = 16;
constexpr int EDIM  = 64;

constexpr int STILE  = 64;              // keys/values per staging step
constexpr int MWAVE  = 16;              // query rows per wave
constexpr int NWAVES = 8;               // 256 threads, wave32
constexpr int MTILE  = MWAVE * NWAVES;  // 128 query rows per block
constexpr int KP     = 72;              // LDS pitch (f16): 144B rows -> 16B-aligned b128
                                        // loads; 36 dwords/row -> lane*36 mod 64 is a
                                        // permutation (conflict-free)
constexpr int NTILES = LQ / STILE;      // 32

__device__ __forceinline__ v8f wmma_f16(v16h a, v16h b, v8f c) {
  return __builtin_amdgcn_wmma_f32_16x16x32_f16(
      false, a, false, b, (short)0, c, false, false);
}

__device__ __forceinline__ v16h cat8(v8h a, v8h b) {
  return __builtin_shufflevector(a, b, 0,1,2,3,4,5,6,7,8,9,10,11,12,13,14,15);
}

// Standard A-operand (16x32 f16): lane's K runs are [8*half,+8) and [16+8*half,+8).
__device__ __forceinline__ v16h load_a_op(const _Float16* row, int half, int kofs) {
  const _Float16* p = row + kofs + half * 8;
  return cat8(*(const v8h*)p, *(const v8h*)(p + 16));
}
// K-permuted A-operand, unused here (the packed B operands are in standard order)
// (swap K blocks [8..15] <-> [16..23]) -> lane reads one contiguous 16-run at 16*half.
__device__ __forceinline__ v16h load_a_perm(const _Float16* row, int half, int kofs) {
  const _Float16* p = row + kofs + half * 16;
  return cat8(*(const v8h*)p, *(const v8h*)(p + 8));
}

__device__ __forceinline__ float relu1(float s) {
  return __builtin_amdgcn_fmed3f(s, 0.0f, 3.0e38f);  // single v_med3_f32
}

__global__ __launch_bounds__(256) void assa_attention_kernel(
    const float* __restrict__ Q, const float* __restrict__ K,
    const float* __restrict__ V, const float* __restrict__ A1,
    const float* __restrict__ A2, float* __restrict__ O)
{
  __shared__ _Float16 ldsK [STILE * KP];   // K tile  [s][e]
  __shared__ _Float16 ldsVt[EDIM  * KP];   // V tile transposed [d][s]

  const int tid  = threadIdx.x;
  const int wave = tid >> 5;
  const int lane = tid & 31;
  const int half = lane >> 4;
  const int lh   = lane & 15;

  const int bid = blockIdx.x;
  const int qt  = bid & (LQ / MTILE - 1);
  const int h   = (bid >> 4) & (NH - 1);
  const int b   = bid >> 8;
  const int qw  = qt * MTILE + wave * MWAVE;

  const float w1 = __expf(A1[0]), w2 = __expf(A2[0]);
  const float alpha1 = w1 / (w1 + w2);
  const float alpha2 = w2 / (w1 + w2);

  // ---- Q^T as B-operand registers (lane = query row), pre-scaled by 1/sqrt(E) ----
  v16h qb0, qb1;
  {
    const float* qp = Q + (((size_t)b * LQ + (qw + lh)) * NH + h) * EDIM;
    #pragma unroll
    for (int i = 0; i < 16; ++i) {
      qb0[i] = (_Float16)(qp[half * 8 + i + (i & 8)]      * 0.125f);  // e in [0,32)
      qb1[i] = (_Float16)(qp[32 + half * 8 + i + (i & 8)] * 0.125f);  // e in [32,64)
    }
  }

  v16h onesA;  // A-operand of all ones: row-sum of P2 via WMMA (softmax denominator)
  #pragma unroll
  for (int i = 0; i < 16; ++i) onesA[i] = (_Float16)1.0f;

  // staging registers (16 K elems + 16 V elems / thread)
  float krx[8], kry[8], vrx[8], vry[8];

  auto load_regs = [&](int s0) {
    #pragma unroll
    for (int it = 0; it < 8; ++it) {
      const int p = it * 256 + tid;
      {
        const int s = p >> 5, e2 = p & 31;   // K: pairs along e (coalesced)
        const float2 kv = *(const float2*)&K[
            (((size_t)b * LQ + s0 + s) * NH + h) * EDIM + 2 * e2];
        krx[it] = kv.x; kry[it] = kv.y;
      }
      {
        const int d = p & 63, s2 = p >> 6;   // V: pairs along s (two coalesced rows)
        const size_t g = (((size_t)b * LQ + s0 + 2 * s2) * NH + h) * EDIM + d;
        vrx[it] = V[g];
        vry[it] = V[g + (size_t)NH * EDIM];
      }
    }
  };

  auto store_regs = [&]() {
    #pragma unroll
    for (int it = 0; it < 8; ++it) {
      const int p = it * 256 + tid;
      {
        const int s = p >> 5, e2 = p & 31;
        h2 t; t[0] = (_Float16)krx[it]; t[1] = (_Float16)kry[it];
        *(h2*)&ldsK[s * KP + 2 * e2] = t;
      }
      {
        const int d = p & 63, s2 = p >> 6;
        h2 t; t[0] = (_Float16)vrx[it]; t[1] = (_Float16)vry[it];
        *(h2*)&ldsVt[d * KP + 2 * s2] = t;   // transposed: contiguous over s
      }
    }
  };

  // outT accumulators: acc[dt] holds out^T rows d = dt*16 + (v + 8*half), col m = lane
  v8f acc1[4], acc2[4], accl = {};
  #pragma unroll
  for (int d = 0; d < 4; ++d) { acc1[d] = {}; acc2[d] = {}; }

  load_regs(0);

  for (int st = 0; st < NTILES; ++st) {
    __syncthreads();                       // previous tile's LDS reads complete
    store_regs();                          // cvt f32->f16, write K/Vt tiles
    if (st + 1 < NTILES) load_regs((st + 1) * STILE);  // overlap HBM with compute
    __syncthreads();

    #pragma unroll
    for (int sc = 0; sc < 2; ++sc) {       // two 32-key chunks per S-tile
      // ---- S^T = K * Q^T : two 16-row score subtiles (s in VGPRs, m in lanes) ----
      v8f c0 = {}, c1 = {};
      {
        const _Float16* k0 = &ldsK[(sc * 32 + lh) * KP];
        c0 = wmma_f16(load_a_op(k0, half, 0),  qb0, c0);
        c0 = wmma_f16(load_a_op(k0, half, 32), qb1, c0);
        const _Float16* k1 = &ldsK[(sc * 32 + 16 + lh) * KP];
        c1 = wmma_f16(load_a_op(k1, half, 0),  qb0, c1);
        c1 = wmma_f16(load_a_op(k1, half, 32), qb1, c1);
      }

      // ---- pack P1 = a1*s*relu(s), P2 = exp(s) straight into B-operand registers.
      // Natural packing gives the standard K order (c0: 8*half+r, c1: 16+8*half+r); the V A-operand
      // below (load_a_op) uses the same order, so the contraction is exact. ----
      v16h p1B, p2B;
      #pragma unroll
      for (int j = 0; j < 4; ++j) {
        {
          const float s0 = c0[2*j], s1 = c0[2*j+1];
          p1B[2*j]   = (_Float16)(relu1(s0) * (alpha1 * s0));
          p1B[2*j+1] = (_Float16)(relu1(s1) * (alpha1 * s1));
          p2B[2*j]   = (_Float16)__expf(s0);
          p2B[2*j+1] = (_Float16)__expf(s1);
        }
        {
          const float s0 = c1[2*j], s1 = c1[2*j+1];
          p1B[8+2*j]   = (_Float16)(relu1(s0) * (alpha1 * s0));
          p1B[8+2*j+1] = (_Float16)(relu1(s1) * (alpha1 * s1));
          p2B[8+2*j]   = (_Float16)__expf(s0);
          p2B[8+2*j+1] = (_Float16)__expf(s1);
        }
      }

      // softmax denominator via ones-matmul (no shuffles, no serial adds)
      accl = wmma_f16(onesA, p2B, accl);

      // ---- out^T += V^T @ P^T : V A-operand shared by both paths ----
      #pragma unroll
      for (int dt = 0; dt < 4; ++dt) {
        const _Float16* vrow = &ldsVt[(dt * 16 + lh) * KP];
        const v16h va = load_a_op(vrow, half, sc * 32);
        acc1[dt] = wmma_f16(va, p1B, acc1[dt]);
        acc2[dt] = wmma_f16(va, p2B, acc2[dt]);
      }
    }
  }

  // ---- epilogue: each lane owns 32 of the 64 outputs of row m = qw + lane%16 ----
  const float s2 = alpha2 / accl[0];
  float* orow = O + (((size_t)b * LQ + (qw + lh)) * NH + h) * EDIM;
  #pragma unroll
  for (int dt = 0; dt < 4; ++dt) {
    float4 o0, o1;
    o0.x = acc1[dt][0] + s2 * acc2[dt][0];
    o0.y = acc1[dt][1] + s2 * acc2[dt][1];
    o0.z = acc1[dt][2] + s2 * acc2[dt][2];
    o0.w = acc1[dt][3] + s2 * acc2[dt][3];
    o1.x = acc1[dt][4] + s2 * acc2[dt][4];
    o1.y = acc1[dt][5] + s2 * acc2[dt][5];
    o1.z = acc1[dt][6] + s2 * acc2[dt][6];
    o1.w = acc1[dt][7] + s2 * acc2[dt][7];
    *(float4*)&orow[dt * 16 + half * 8]     = o0;
    *(float4*)&orow[dt * 16 + half * 8 + 4] = o1;
  }
}

extern "C" void kernel_launch(void* const* d_in, const int* in_sizes, int n_in,
                              void* d_out, int out_size, void* d_ws, size_t ws_size,
                              hipStream_t stream) {
  (void)in_sizes; (void)n_in; (void)out_size; (void)d_ws; (void)ws_size;
  const float* Q  = (const float*)d_in[0];
  const float* K  = (const float*)d_in[1];
  const float* V  = (const float*)d_in[2];
  const float* a1 = (const float*)d_in[3];
  const float* a2 = (const float*)d_in[4];
  float* O = (float*)d_out;

  const int grid = BATCH * NH * (LQ / MTILE);  // 512 blocks
  hipLaunchKernelGGL(assa_attention_kernel, dim3(grid), dim3(256), 0, stream,
                     Q, K, V, a1, a2, O);
}
